// GetDensity_13932873908300
// MI455X (gfx1250) — matrix operand order repaired and checked by reading, not compiled, not run on hardware
//
#include <hip/hip_runtime.h>
#include <math.h>

typedef __attribute__((ext_vector_type(16))) _Float16 v16h;
typedef __attribute__((ext_vector_type(8)))  _Float16 v8h;
typedef __attribute__((ext_vector_type(8)))  float    v8f;

#define NWAVE   8
#define PORI    13
#define DFEAT   24
#define HIDDEN  128
#define DPAD    32              // density row stride (K padded 24 -> 32)
// pi / CUTOFF (CUTOFF = 5.0)
#define PI_OVER_CUT 0.62831853071795864769f

// ---------------------------------------------------------------------------
// zero a float buffer
// ---------------------------------------------------------------------------
__global__ void zero_f32(float* __restrict__ p, int n) {
    int i = blockIdx.x * blockDim.x + threadIdx.x;
    if (i < n) p[i] = 0.0f;
}

// ---------------------------------------------------------------------------
// Edge pass: radial[e,p,w] = ang*fc*gauss (optionally * per-neighbor message),
// atomically accumulated into orbital[center]. The 8.3MB orbital table is
// L2-resident on MI455X (192MB L2) so all atomic RMW traffic stays on-chip.
// ---------------------------------------------------------------------------
__global__ void edge_accum(const float* __restrict__ cart,
                           const int*   __restrict__ center,
                           const int*   __restrict__ neigh,
                           const float* __restrict__ shifts,
                           const int*   __restrict__ species,
                           const float* __restrict__ rs,
                           const float* __restrict__ inta,
                           const float* __restrict__ params,
                           const float* __restrict__ msg,     // [A,24] or null
                           float*       __restrict__ orbital, // [A,13,8]
                           int nedge) {
    int e = blockIdx.x * blockDim.x + threadIdx.x;
    if (e >= nedge) return;

    int c  = center[e];
    int n  = neigh[e];
    float x = cart[3*c+0] - cart[3*n+0] - shifts[3*e+0];
    float y = cart[3*c+1] - cart[3*n+1] - shifts[3*e+1];
    float z = cart[3*c+2] - cart[3*n+2] - shifts[3*e+2];
    float d = sqrtf(x*x + y*y + z*z);

    float t  = 0.5f * __cosf(d * PI_OVER_CUT) + 0.5f;
    float fc = t * t;

    int sp = species[n];
    float g[NWAVE];
#pragma unroll
    for (int w = 0; w < NWAVE; ++w) {
        float dr = d - rs[sp*NWAVE + w];
        g[w] = __expf(-inta[sp*NWAVE + w] * dr * dr) * params[sp*NWAVE + w];
    }

    float ang[PORI] = {1.0f, x, y, z,
                       x*x, x*y, x*z,
                       y*x, y*y, y*z,
                       z*x, z*y, z*z};

#pragma unroll
    for (int p = 0; p < PORI; ++p) {
        int   ip   = (p == 0) ? 0 : ((p < 4) ? 1 : 2);
        float base = ang[p] * fc;
#pragma unroll
        for (int w = 0; w < NWAVE; ++w) {
            float r = base * g[w];
            if (msg) r *= msg[n*DFEAT + ip*NWAVE + w];
            atomicAdd(&orbital[c*(PORI*NWAVE) + p*NWAVE + w], r);
        }
    }
}

// ---------------------------------------------------------------------------
// dens[a*stride + i*8+w] = sum_{p: index_para[p]==i} orbital[a,p,w]^2
// Columns [DFEAT, stride) are zero-filled (K padding for WMMA).
// ---------------------------------------------------------------------------
__global__ void square_contract(const float* __restrict__ orbital,
                                float*       __restrict__ dens,
                                int natom, int stride) {
    int tid = blockIdx.x * blockDim.x + threadIdx.x;
    if (tid >= natom * stride) return;
    int a  = tid / stride;
    int iw = tid % stride;
    if (iw >= DFEAT) { dens[tid] = 0.0f; return; }
    int i  = iw / NWAVE;
    int w  = iw % NWAVE;
    int p0, p1;
    if      (i == 0) { p0 = 0; p1 = 1;  }
    else if (i == 1) { p0 = 1; p1 = 4;  }
    else             { p0 = 4; p1 = 13; }
    float s = 0.0f;
    for (int p = p0; p < p1; ++p) {
        float v = orbital[a*(PORI*NWAVE) + p*NWAVE + w];
        s += v * v;
    }
    dens[tid] = s;
}

// ---------------------------------------------------------------------------
// Pre-swizzle w1 [24,128] f32 into GEMM1 B-fragment layout:
//   dst[tile(8)][lane(32)][j(16)] f16, k = (j<8 ? half*8+j : 16+half*8+j-8) (zero for k>=24), n = tile*16+l16
// ---------------------------------------------------------------------------
__global__ void swizzle_w1(const float* __restrict__ w1, _Float16* __restrict__ dst) {
    int t = blockIdx.x * blockDim.x + threadIdx.x;
    if (t >= 8*32*16) return;
    int j    = t & 15;
    int lane = (t >> 4) & 31;
    int tile = t >> 9;
    int half = lane >> 4, l16 = lane & 15;
    int n = tile*16 + l16;
    int k = (j < 8) ? (half*8 + j) : (16 + half*8 + (j - 8));
    dst[t] = (k < DFEAT) ? (_Float16)w1[k*HIDDEN + n] : (_Float16)0.0f;
}

// ---------------------------------------------------------------------------
// Pre-swizzle w2 [128,24] f32 into GEMM2 B-fragment layout:
//   dst[c0(4)][t2(2)][lane(32)][j(16)] f16, k = c0*32+(j<8 ? half*8+j : 16+half*8+j-8), n = t2*16+l16
// ---------------------------------------------------------------------------
__global__ void swizzle_w2(const float* __restrict__ w2, _Float16* __restrict__ dst) {
    int t = blockIdx.x * blockDim.x + threadIdx.x;
    if (t >= 4*2*32*16) return;
    int j    = t & 15;
    int lane = (t >> 4) & 31;
    int frag = t >> 9;                  // c0*2 + t2
    int t2   = frag & 1;
    int c0   = frag >> 1;
    int half = lane >> 4, l16 = lane & 15;
    int n = t2*16 + l16;
    int k = c0*32 + ((j < 8) ? (half*8 + j) : (16 + half*8 + (j - 8)));
    dst[t] = (n < DFEAT) ? (_Float16)w2[k*DFEAT + n] : (_Float16)0.0f;
}

// ---------------------------------------------------------------------------
// Pad b2 [24] -> [32] with zeros
// ---------------------------------------------------------------------------
__global__ void pad_b2(const float* __restrict__ b2, float* __restrict__ dst) {
    int i = threadIdx.x;
    if (i < 32) dst[i] = (i < DFEAT) ? b2[i] : 0.0f;
}

// ---------------------------------------------------------------------------
// MLP: out = tanh(dens @ w1 + b1) @ w2 + b2 via v_wmma_f32_16x16x32_f16.
// One wave = one 16-atom tile. All fragment loads are branch-free vector
// loads (density is K-padded to 32; weights pre-swizzled into fragments).
// Hidden tile transposed through LDS (C/D layout -> A layout).
// ---------------------------------------------------------------------------
__global__ __launch_bounds__(256)
void mlp_wmma(const float*    __restrict__ dens,  // [A,32] padded
              const _Float16* __restrict__ w1f,   // [8][32][16] fragments
              const float*    __restrict__ b1,    // [128]
              const _Float16* __restrict__ w2f,   // [8][32][16] fragments
              const float*    __restrict__ b2p,   // [32] padded
              float*          __restrict__ out,   // [A,24]
              int natom) {
    __shared__ __align__(32) _Float16 hbuf[8][16][HIDDEN];  // 32KB/block

    int lane = threadIdx.x & 31;
    int wv   = threadIdx.x >> 5;
    int tile = blockIdx.x * 8 + wv;
    int ntile = natom / 16;
    if (tile >= ntile) return;          // uniform per wave: EXEC stays all-1

    int half = lane >> 4;
    int l16  = lane & 15;

    // ---- A fragment: 16 density rows, branch-free float4 loads + cvt ----
    int m = tile*16 + l16;
    const float4* dr = (const float4*)(dens + (size_t)m * DPAD);
    float4 f0 = dr[half*2 + 0];
    float4 f1 = dr[half*2 + 1];
    float4 f2 = dr[half*2 + 4];
    float4 f3 = dr[half*2 + 5];
    v16h afrag;
    afrag[0]=(_Float16)f0.x;  afrag[1]=(_Float16)f0.y;
    afrag[2]=(_Float16)f0.z;  afrag[3]=(_Float16)f0.w;
    afrag[4]=(_Float16)f1.x;  afrag[5]=(_Float16)f1.y;
    afrag[6]=(_Float16)f1.z;  afrag[7]=(_Float16)f1.w;
    afrag[8]=(_Float16)f2.x;  afrag[9]=(_Float16)f2.y;
    afrag[10]=(_Float16)f2.z; afrag[11]=(_Float16)f2.w;
    afrag[12]=(_Float16)f3.x; afrag[13]=(_Float16)f3.y;
    afrag[14]=(_Float16)f3.z; afrag[15]=(_Float16)f3.w;

    const v16h* w1v = (const v16h*)w1f;
    const v16h* w2v = (const v16h*)w2f;

    // ---- GEMM1: hidden = dens @ w1 + b1 (8 N-tiles, one K=32 WMMA each) ----
    v8f acc[8];
#pragma unroll
    for (int t = 0; t < 8; ++t) {
        float bias  = b1[t*16 + l16];
        v16h  bfrag = w1v[t*32 + lane];
        v8f   cfr;
#pragma unroll
        for (int i = 0; i < 8; ++i) cfr[i] = bias;
        acc[t] = __builtin_amdgcn_wmma_f32_16x16x32_f16(
                     false, afrag, false, bfrag, (short)0, cfr, false, false);
    }

    // ---- tanh + transpose through LDS (C/D layout -> [m][n]) ----
#pragma unroll
    for (int t = 0; t < 8; ++t) {
#pragma unroll
        for (int i = 0; i < 8; ++i) {
            hbuf[wv][i + 8*half][t*16 + l16] = (_Float16)tanhf(acc[t][i]);
        }
    }

    // ---- GEMM2: out = hidden @ w2 + b2 (4 K-chunks x 2 N-tiles) ----
    v8f acc2[2];
#pragma unroll
    for (int t2 = 0; t2 < 2; ++t2) {
        float bias = b2p[t2*16 + l16];
#pragma unroll
        for (int i = 0; i < 8; ++i) acc2[t2][i] = bias;
    }
#pragma unroll
    for (int c0 = 0; c0 < 4; ++c0) {
        int kb = c0 * 32;
        v8h lo = *(const v8h*)&hbuf[wv][l16][kb + half*8];
        v8h hi = *(const v8h*)&hbuf[wv][l16][kb + 16 + half*8];
        v16h a2 = __builtin_shufflevector(lo, hi,
                      0,1,2,3,4,5,6,7,8,9,10,11,12,13,14,15);
#pragma unroll
        for (int t2 = 0; t2 < 2; ++t2) {
            v16h bfrag = w2v[(c0*2 + t2)*32 + lane];
            acc2[t2] = __builtin_amdgcn_wmma_f32_16x16x32_f16(
                           false, a2, false, bfrag, (short)0, acc2[t2], false, false);
        }
    }

    // ---- store (only valid N columns; WMMAs all done, divergence OK) ----
#pragma unroll
    for (int t2 = 0; t2 < 2; ++t2) {
        int nn = t2*16 + l16;
        if (nn < DFEAT) {
#pragma unroll
            for (int i = 0; i < 8; ++i) {
                int a = tile*16 + i + 8*half;
                out[a*DFEAT + nn] = acc2[t2][i];
            }
        }
    }
}

// ---------------------------------------------------------------------------
extern "C" void kernel_launch(void* const* d_in, const int* in_sizes, int n_in,
                              void* d_out, int out_size, void* d_ws, size_t ws_size,
                              hipStream_t stream) {
    const float* cart   = (const float*)d_in[0];
    const int*   nl     = (const int*)  d_in[1];
    const float* shifts = (const float*)d_in[2];
    const int*   spec   = (const int*)  d_in[3];
    const float* rs     = (const float*)d_in[4];
    const float* inta   = (const float*)d_in[5];
    const float* params = (const float*)d_in[6];
    const float* w1[2]  = {(const float*)d_in[7],  (const float*)d_in[11]};
    const float* b1[2]  = {(const float*)d_in[8],  (const float*)d_in[12]};
    const float* w2[2]  = {(const float*)d_in[9],  (const float*)d_in[13]};
    const float* b2[2]  = {(const float*)d_in[10], (const float*)d_in[14]};

    const int nedge = in_sizes[1] / 2;
    const int natom = in_sizes[3];
    const int* center = nl;
    const int* neigh  = nl + nedge;

    // workspace layout (256B-aligned chunks)
    char* ws = (char*)d_ws;
    size_t off = 0;
    auto take = [&](size_t bytes) -> char* {
        char* p = ws + off;
        off += (bytes + 255) & ~(size_t)255;
        return p;
    };
    float*     orbital = (float*)    take((size_t)natom * PORI * NWAVE * 4);
    float*     dens32  = (float*)    take((size_t)natom * DPAD * 4);
    float*     msg     = (float*)    take((size_t)natom * DFEAT * 4);
    float*     b2p0    = (float*)    take(32 * 4);
    float*     b2p1    = (float*)    take(32 * 4);
    _Float16*  w1f0    = (_Float16*) take(8*32*16 * 2);
    _Float16*  w1f1    = (_Float16*) take(8*32*16 * 2);
    _Float16*  w2f0    = (_Float16*) take(8*32*16 * 2);
    _Float16*  w2f1    = (_Float16*) take(8*32*16 * 2);
    _Float16*  w1f[2]  = {w1f0, w1f1};
    _Float16*  w2f[2]  = {w2f0, w2f1};
    float*     b2p[2]  = {b2p0, b2p1};
    float*     outF    = (float*)d_out;

    const int B      = 256;
    const int norb   = natom * PORI * NWAVE;
    const int gZero  = (norb + B - 1) / B;
    const int gEdge  = (nedge + B - 1) / B;
    const int gDensP = (natom * DPAD + B - 1) / B;
    const int gDensO = (natom * DFEAT + B - 1) / B;
    const int ntile  = natom / 16;
    const int gMlp   = (ntile + 7) / 8;
    const int gSwz   = (8*32*16 + B - 1) / B;

    // prep: swizzle weights into fragment layout, pad biases
    for (int it = 0; it < 2; ++it) {
        swizzle_w1<<<gSwz, B, 0, stream>>>(w1[it], w1f[it]);
        swizzle_w2<<<gSwz, B, 0, stream>>>(w2[it], w2f[it]);
        pad_b2<<<1, 32, 0, stream>>>(b2[it], b2p[it]);
    }

    // pass 0: orbital = segsum(radial); density = contract(orbital^2)
    zero_f32<<<gZero, B, 0, stream>>>(orbital, norb);
    edge_accum<<<gEdge, B, 0, stream>>>(cart, center, neigh, shifts, spec,
                                        rs, inta, params, nullptr, orbital, nedge);
    square_contract<<<gDensP, B, 0, stream>>>(orbital, dens32, natom, DPAD);

    // two message-passing iterations
    for (int it = 0; it < 2; ++it) {
        mlp_wmma<<<gMlp, B, 0, stream>>>(dens32, w1f[it], b1[it], w2f[it],
                                         b2p[it], msg, natom);
        zero_f32<<<gZero, B, 0, stream>>>(orbital, norb);
        edge_accum<<<gEdge, B, 0, stream>>>(cart, center, neigh, shifts, spec,
                                            rs, inta, params, msg, orbital, nedge);
        if (it == 1) {
            square_contract<<<gDensO, B, 0, stream>>>(orbital, outF, natom, DFEAT);
        } else {
            square_contract<<<gDensP, B, 0, stream>>>(orbital, dens32, natom, DPAD);
        }
    }
}
